// EGATLayer_41893111005429
// MI455X (gfx1250) — compile-verified, matrix operand order checked by reading, not run on hardware
//
#include <hip/hip_runtime.h>
#include <hip/hip_bf16.h>

typedef _Float16 half_t;
typedef __attribute__((ext_vector_type(16))) _Float16 v16h;
typedef __attribute__((ext_vector_type(8)))  _Float16 v8h;
typedef __attribute__((ext_vector_type(8)))  float    v8f;

// ---------------------------------------------------------------------------
// WMMA helpers (CDNA5 wave32, V_WMMA_F32_16X16X32_F16)
// A fragment: lane L holds row M=L&15; halves [0..7] = K kbase..kbase+7,
// halves [8..15] = K kbase+16..kbase+23, kbase = (L>=16)*8.  B symmetric with
// columns stored K-contiguous (i.e. B^T row-major).  Two 16B loads per frag.
// ---------------------------------------------------------------------------
__device__ __forceinline__ v16h load_frag(const half_t* p0, int stride) {
  int lane = threadIdx.x & 31;
  const half_t* p = p0 + (lane & 15) * stride + ((lane >> 4) << 3);
  v8h lo = *(const v8h*)(p);
  v8h hi = *(const v8h*)(p + 16);
  v16h r;
#pragma unroll
  for (int i = 0; i < 8; ++i) { r[i] = lo[i]; r[i + 8] = hi[i]; }
  return r;
}

__device__ __forceinline__ v8f wmma32(v16h a, v16h b, v8f c) {
  return __builtin_amdgcn_wmma_f32_16x16x32_f16(false, a, false, b,
                                                (short)0, c, false, false);
}

// ---------------------------------------------------------------------------
// Kernel 0: transpose + convert weights to f16 in workspace
// W_[64,256] -> Wt[256][64] ; W_mlp[256,64] -> Wt[64][256]
// ---------------------------------------------------------------------------
__global__ void prep_kernel(const float* __restrict__ W_ni, const float* __restrict__ W_nj,
                            const float* __restrict__ W_fij, const float* __restrict__ W_src,
                            const float* __restrict__ W_mlpn, const float* __restrict__ W_mlpe,
                            half_t* __restrict__ Wt_ni, half_t* __restrict__ Wt_nj,
                            half_t* __restrict__ Wt_fij, half_t* __restrict__ Wt_src,
                            half_t* __restrict__ Wt_mlpn, half_t* __restrict__ Wt_mlpe) {
  int tid = blockIdx.x * blockDim.x + threadIdx.x;
  int nt = gridDim.x * blockDim.x;
  for (int i = tid; i < 64 * 256; i += nt) {
    int k = i >> 8, c = i & 255;
    Wt_ni[c * 64 + k]  = (half_t)W_ni[i];
    Wt_nj[c * 64 + k]  = (half_t)W_nj[i];
    Wt_fij[c * 64 + k] = (half_t)W_fij[i];
    Wt_src[c * 64 + k] = (half_t)W_src[i];
  }
  for (int i = tid; i < 256 * 64; i += nt) {
    int k = i >> 6, c = i & 63;
    Wt_mlpn[c * 256 + k] = (half_t)W_mlpn[i];
    Wt_mlpe[c * 256 + k] = (half_t)W_mlpe[i];
  }
}

// Kernel 1: zero h_out, init segment max (encoded) / denom
__global__ void init_kernel(float* __restrict__ h_out, unsigned* __restrict__ node_max,
                            float* __restrict__ node_denom, int N) {
  size_t idx = (size_t)blockIdx.x * 256 + threadIdx.x;
  if (idx < (size_t)N * 256) h_out[idx] = 0.f;
  if (idx < (size_t)N * 4) { node_max[idx] = 0x00800000u; node_denom[idx] = 0.f; }
}

// ---------------------------------------------------------------------------
// Kernel 2: P_ni / P_nj / P_src = nfeats @ W_{ni,nj,src}   [N,256] f32 each
// 128 threads (4 waves) per 16-row tile; wave w owns col-blocks w*4..w*4+3
// ---------------------------------------------------------------------------
__global__ void __launch_bounds__(128)
node_proj_kernel(const float* __restrict__ nfeats,
                 const half_t* __restrict__ Wt_ni, const half_t* __restrict__ Wt_nj,
                 const half_t* __restrict__ Wt_src,
                 float* __restrict__ Pni, float* __restrict__ Pnj, float* __restrict__ Psrc) {
  __shared__ half_t sA[16 * 64];
  int tid = threadIdx.x;
  int row0 = blockIdx.x * 16;
  for (int i = tid; i < 16 * 64; i += 128)
    sA[i] = (half_t)nfeats[(size_t)row0 * 64 + i];
  __syncthreads();

  int w = tid >> 5, lane = tid & 31, n = lane & 15, hb = lane >> 4;
  v16h a0 = load_frag(sA, 64);
  v16h a1 = load_frag(sA + 32, 64);
  const half_t* Wts[3] = {Wt_ni, Wt_nj, Wt_src};
  float* Ps[3] = {Pni, Pnj, Psrc};
#pragma unroll
  for (int m = 0; m < 3; ++m) {
#pragma unroll
    for (int cb = 0; cb < 4; ++cb) {
      int col0 = (w * 4 + cb) * 16;
      v16h b0 = load_frag(Wts[m] + col0 * 64, 64);
      v16h b1 = load_frag(Wts[m] + col0 * 64 + 32, 64);
      v8f c = {};
      c = wmma32(a0, b0, c);
      c = wmma32(a1, b1, c);
      float* out = Ps[m] + (size_t)row0 * 256 + col0;
#pragma unroll
      for (int r = 0; r < 8; ++r)
        out[(size_t)(r + 8 * hb) * 256 + n] = c[r];
    }
  }
}

// ---------------------------------------------------------------------------
// Kernel 3 (fused edge stage): f_fij WMMA + gather(P_ni[src],P_nj[dst]) +
// leakyReLU -> f16 LDS tile; attention logits + atomicMax(dst); edge-MLP
// WMMA (K=256) + ELU -> edge output.  128 threads / 16-edge tile.
// ---------------------------------------------------------------------------
__global__ void __launch_bounds__(128)
edge_stage1_kernel(const float* __restrict__ efeats,
                   const int* __restrict__ src, const int* __restrict__ dst,
                   const float* __restrict__ Pni, const float* __restrict__ Pnj,
                   const half_t* __restrict__ Wt_fij, const half_t* __restrict__ Wt_mlpe,
                   const float* __restrict__ bias_e, const float* __restrict__ attn,
                   const float* __restrict__ b_mlp_edge,
                   float* __restrict__ e_buf, unsigned* __restrict__ node_max,
                   float* __restrict__ edge_out) {
  __shared__ half_t sE[16 * 64];
  __shared__ half_t sF[16 * 256];
  __shared__ float  s_attn[256];
  __shared__ int    s_src[16], s_dst[16];
  int tid = threadIdx.x;
  int e0 = blockIdx.x * 16;

  for (int i = tid; i < 16 * 64; i += 128)
    sE[i] = (half_t)efeats[(size_t)e0 * 64 + i];
  for (int i = tid; i < 256; i += 128) s_attn[i] = attn[i];
  if (tid < 16) { s_src[tid] = src[e0 + tid]; s_dst[tid] = dst[e0 + tid]; }
  __syncthreads();

  int w = tid >> 5, lane = tid & 31, n = lane & 15, hb = lane >> 4;
  v16h a0 = load_frag(sE, 64);
  v16h a1 = load_frag(sE + 32, 64);
#pragma unroll
  for (int cb = 0; cb < 4; ++cb) {
    int col0 = (w * 4 + cb) * 16;
    v16h b0 = load_frag(Wt_fij + col0 * 64, 64);
    v16h b1 = load_frag(Wt_fij + col0 * 64 + 32, 64);
    v8f c = {};
    c = wmma32(a0, b0, c);
    c = wmma32(a1, b1, c);
    int col = col0 + n;
    float bv = bias_e[col];
#pragma unroll
    for (int r = 0; r < 8; ++r) {
      int em = r + 8 * hb;
      float v = c[r] + Pni[(size_t)s_src[em] * 256 + col]
                     + Pnj[(size_t)s_dst[em] * 256 + col] + bv;
      v = (v >= 0.f) ? v : 0.01f * v;               // leaky_relu(0.01)
      sF[em * 256 + col] = (half_t)v;
    }
  }
  __syncthreads();

  // attention logits e[eh] = sum_d f_out[e,h,d]*attn[h,d]; segment-max(dst)
  if (tid < 64) {
    int em = tid >> 2, h = tid & 3;
    float acc = 0.f;
#pragma unroll 8
    for (int d = 0; d < 64; ++d)
      acc += (float)sF[em * 256 + h * 64 + d] * s_attn[h * 64 + d];
    e_buf[(size_t)(e0 + em) * 4 + h] = acc;
    unsigned u = __float_as_uint(acc);
    u = (u & 0x80000000u) ? ~u : (u | 0x80000000u);  // order-preserving key
    atomicMax(&node_max[(size_t)s_dst[em] * 4 + h], u);
  }

  // edge MLP: elu(f_out @ W_mlp_edge + b); wave w -> output col block w
  {
    int col0 = w * 16;
    v8f c = {};
#pragma unroll
    for (int ks = 0; ks < 8; ++ks) {
      v16h a = load_frag(sF + ks * 32, 256);
      v16h b = load_frag(Wt_mlpe + col0 * 256 + ks * 32, 256);
      c = wmma32(a, b, c);
    }
    int col = col0 + n;
    float bv = b_mlp_edge[col];
#pragma unroll
    for (int r = 0; r < 8; ++r) {
      float v = c[r] + bv;
      v = (v > 0.f) ? v : (expf(v) - 1.f);          // elu
      edge_out[(size_t)(e0 + r + 8 * hb) * 64 + col] = v;
    }
  }
}

// Kernel 4: e_exp = exp(e - max[dst]);  denom[dst] += e_exp
__global__ void __launch_bounds__(256)
edge_softmax_kernel(const int* __restrict__ dst, const unsigned* __restrict__ node_max,
                    float* __restrict__ e_buf, float* __restrict__ node_denom, int E) {
  size_t idx = (size_t)blockIdx.x * 256 + threadIdx.x;
  if (idx >= (size_t)E * 4) return;
  int e = (int)(idx >> 2), h = (int)(idx & 3);
  int d = dst[e];
  unsigned u = node_max[(size_t)d * 4 + h];
  u = (u & 0x80000000u) ? (u ^ 0x80000000u) : ~u;   // decode key -> float
  float m = __uint_as_float(u);
  float ex = expf(e_buf[idx] - m);
  e_buf[idx] = ex;
  atomicAdd(&node_denom[(size_t)d * 4 + h], ex);
}

// Kernel 5: h_out[dst] += P_src[src] * a   (coalesced f32 atomics)
__global__ void __launch_bounds__(256)
scatter_kernel(const int* __restrict__ src, const int* __restrict__ dst,
               const float* __restrict__ Psrc, const float* __restrict__ e_buf,
               const float* __restrict__ node_denom, float* __restrict__ h_out) {
  int tid = threadIdx.x;           // tid = column 0..255, head = tid>>6
  int e0 = blockIdx.x * 16;
  int h = tid >> 6;
  for (int i = 0; i < 16; ++i) {
    int e = e0 + i;
    int s = src[e], d = dst[e];
    float a = e_buf[(size_t)e * 4 + h] / node_denom[(size_t)d * 4 + h];
    float v = Psrc[(size_t)s * 256 + tid] * a;
    atomicAdd(&h_out[(size_t)d * 256 + tid], v);
  }
}

// Kernel 6: node = elu(h_out @ W_mlp_node + b)
__global__ void __launch_bounds__(128)
node_mlp_kernel(const float* __restrict__ h_out, const half_t* __restrict__ Wt_mlpn,
                const float* __restrict__ b_mlp_node, float* __restrict__ node_out) {
  __shared__ half_t sH[16 * 256];
  int tid = threadIdx.x;
  int row0 = blockIdx.x * 16;
  for (int i = tid; i < 16 * 256; i += 128)
    sH[i] = (half_t)h_out[(size_t)row0 * 256 + i];
  __syncthreads();

  int w = tid >> 5, lane = tid & 31, n = lane & 15, hb = lane >> 4;
  int col0 = w * 16;
  v8f c = {};
#pragma unroll
  for (int ks = 0; ks < 8; ++ks) {
    v16h a = load_frag(sH + ks * 32, 256);
    v16h b = load_frag(Wt_mlpn + col0 * 256 + ks * 32, 256);
    c = wmma32(a, b, c);
  }
  int col = col0 + n;
  float bv = b_mlp_node[col];
#pragma unroll
  for (int r = 0; r < 8; ++r) {
    float v = c[r] + bv;
    v = (v > 0.f) ? v : (expf(v) - 1.f);
    node_out[(size_t)(row0 + r + 8 * hb) * 64 + col] = v;
  }
}

// ---------------------------------------------------------------------------
extern "C" void kernel_launch(void* const* d_in, const int* in_sizes, int n_in,
                              void* d_out, int out_size, void* d_ws, size_t ws_size,
                              hipStream_t stream) {
  const float* nfeats     = (const float*)d_in[0];
  const float* efeats     = (const float*)d_in[1];
  const int*   src        = (const int*)d_in[2];
  const int*   dst        = (const int*)d_in[3];
  const float* W_ni       = (const float*)d_in[4];
  const float* W_nj       = (const float*)d_in[5];
  const float* W_fij      = (const float*)d_in[6];
  const float* bias_e     = (const float*)d_in[7];
  const float* attn       = (const float*)d_in[8];
  const float* W_src      = (const float*)d_in[9];
  const float* W_mlp_node = (const float*)d_in[10];
  const float* b_mlp_node = (const float*)d_in[11];
  const float* W_mlp_edge = (const float*)d_in[12];
  const float* b_mlp_edge = (const float*)d_in[13];

  const int N = in_sizes[0] / 64;   // 50000 (divisible by 16)
  const int E = in_sizes[1] / 64;   // 800000 (divisible by 16)

  char* ws = (char*)d_ws;
  size_t off = 0;
  auto carve = [&](size_t bytes) -> void* {
    void* p = ws + off;
    off += (bytes + 255) & ~(size_t)255;
    return p;
  };
  half_t* Wt_ni   = (half_t*)carve(256 * 64 * sizeof(half_t));
  half_t* Wt_nj   = (half_t*)carve(256 * 64 * sizeof(half_t));
  half_t* Wt_fij  = (half_t*)carve(256 * 64 * sizeof(half_t));
  half_t* Wt_src  = (half_t*)carve(256 * 64 * sizeof(half_t));
  half_t* Wt_mlpn = (half_t*)carve(64 * 256 * sizeof(half_t));
  half_t* Wt_mlpe = (half_t*)carve(64 * 256 * sizeof(half_t));
  float*  Pni     = (float*)carve((size_t)N * 256 * sizeof(float));
  float*  Pnj     = (float*)carve((size_t)N * 256 * sizeof(float));
  float*  Psrc    = (float*)carve((size_t)N * 256 * sizeof(float));
  float*  e_buf   = (float*)carve((size_t)E * 4 * sizeof(float));
  unsigned* node_max   = (unsigned*)carve((size_t)N * 4 * sizeof(unsigned));
  float*  node_denom   = (float*)carve((size_t)N * 4 * sizeof(float));
  float*  h_out        = (float*)carve((size_t)N * 256 * sizeof(float));

  float* node_out = (float*)d_out;
  float* edge_out = (float*)d_out + (size_t)N * 64;

  prep_kernel<<<64, 256, 0, stream>>>(W_ni, W_nj, W_fij, W_src, W_mlp_node, W_mlp_edge,
                                      Wt_ni, Wt_nj, Wt_fij, Wt_src, Wt_mlpn, Wt_mlpe);
  size_t initN = (size_t)N * 256;
  init_kernel<<<(unsigned)((initN + 255) / 256), 256, 0, stream>>>(h_out, node_max,
                                                                   node_denom, N);
  node_proj_kernel<<<N / 16, 128, 0, stream>>>(nfeats, Wt_ni, Wt_nj, Wt_src,
                                               Pni, Pnj, Psrc);
  edge_stage1_kernel<<<E / 16, 128, 0, stream>>>(efeats, src, dst, Pni, Pnj,
                                                 Wt_fij, Wt_mlpe, bias_e, attn,
                                                 b_mlp_edge, e_buf, node_max, edge_out);
  edge_softmax_kernel<<<(unsigned)(((size_t)E * 4 + 255) / 256), 256, 0, stream>>>(
      dst, node_max, e_buf, node_denom, E);
  scatter_kernel<<<E / 16, 256, 0, stream>>>(src, dst, Psrc, e_buf, node_denom, h_out);
  node_mlp_kernel<<<N / 16, 128, 0, stream>>>(h_out, Wt_mlpn, b_mlp_node, node_out);
}
